// Attention_31610959299247
// MI455X (gfx1250) — compile-verified, matrix operand order checked by reading, not run on hardware
//
#include <hip/hip_runtime.h>
#include <hip/hip_bf16.h>

// ---------------- problem constants ----------------
#define HD      128            // head dim
#define NH      32             // heads
#define SEQ     8192           // cache length
#define NLAYERS 32
#define NCHUNK  16             // flash-decoding chunks per head
#define CHUNK   (SEQ / NCHUNK) // 512 keys per block
#define TILE    128            // keys staged per LDS tile
#define NTILE   (CHUNK / TILE) // 4
#define TILE_ELEMS (TILE * HD) // 16384 bf16 = 32KB

typedef __attribute__((ext_vector_type(16))) __bf16 v16bf;
typedef __attribute__((ext_vector_type(8)))  __bf16 v8bf;
typedef __attribute__((ext_vector_type(4)))  __bf16 v4bf;
typedef __attribute__((ext_vector_type(8)))  float  v8f;
typedef __attribute__((ext_vector_type(4)))  float  v4f;
typedef __attribute__((ext_vector_type(4)))  unsigned int u32x4;
typedef __attribute__((ext_vector_type(8)))  int    i32x8;
typedef __attribute__((ext_vector_type(4)))  int    i32x4;

// ---- CDNA5 async data-movement feature detection (device pass only) ----
#if defined(__gfx1250__) && __has_builtin(__builtin_amdgcn_tensor_load_to_lds) && __has_builtin(__builtin_amdgcn_s_wait_tensorcnt)
#  define HAVE_TDM 1
#else
#  define HAVE_TDM 0
#endif
#if !HAVE_TDM && defined(__gfx1250__) && __has_builtin(__builtin_amdgcn_global_load_async_to_lds_b128)
#  define HAVE_ASYNC 1
#else
#  define HAVE_ASYNC 0
#endif

// ---------------- fp32 -> bf16 KV cache conversion (one-time, 256MB read / 128MB write) ----
__global__ __launch_bounds__(256)
void convert_kv_kernel(const float* __restrict__ K, const float* __restrict__ V,
                       void* __restrict__ Kbp, void* __restrict__ Vbp, long n4) {
    v4bf* __restrict__ Kb4 = (v4bf*)Kbp;
    v4bf* __restrict__ Vb4 = (v4bf*)Vbp;
    const v4f* __restrict__ K4 = (const v4f*)K;
    const v4f* __restrict__ V4 = (const v4f*)V;
    long i = (long)blockIdx.x * blockDim.x + threadIdx.x;
    long stride = (long)gridDim.x * blockDim.x;
    for (; i < n4; i += stride) {
        v4f k = K4[i], v = V4[i];
        v4bf kb, vb;
        #pragma unroll
        for (int j = 0; j < 4; ++j) { kb[j] = (__bf16)k[j]; vb[j] = (__bf16)v[j]; }
        Kb4[i] = kb; Vb4[i] = vb;
    }
}

// ---------------- tile staging: TDM -> async-to-LDS -> manual copy ------------------------
// BF=true : src is bf16, one contiguous 32KB tile -> DMA / async copy
// BF=false: src is fp32, convert in-register while staging (manual only)
template<bool BF>
__device__ __forceinline__ void stage_issue(__bf16* dst, const void* srcv, int tid) {
    if (BF) {
#if HAVE_TDM
        // Single-wave issue: TDM ignores EXEC; one descriptor moves the whole 32KB tile.
        if (tid < 32) {
            unsigned ldsa = (unsigned)(unsigned long long)
                            (__attribute__((address_space(3))) char*)dst;
            unsigned long long ga = (unsigned long long)srcv;
            // D# group0: count=1 | lds_addr | global_addr[56:0] | type=2
            u32x4 g0 = { 1u, ldsa, (unsigned)ga,
                         (unsigned)((ga >> 32) & 0x01FFFFFFu) | (2u << 30) };
            // D# group1: data_size=2B; tensor_dim0=16384; tensor_dim1=1;
            //            tile_dim0=16384 (1-D tile); tensor_dim0_stride=16384
            i32x8 g1 = { (int)0x00010000,   // data_size=1 (2 bytes) @ bits17:16
                         (int)0x40000000,   // tensor_dim0[15:0]=16384 @ bits63:48
                         (int)0x00010000,   // tensor_dim1[15:0]=1    @ bits95:80
                         (int)0x40000000,   // tile_dim0=16384        @ bits127:112
                         0,                 // tile_dim1=0 (unused), tile_dim2=0
                         16384,             // tensor_dim0_stride[31:0]
                         0, 0 };
            i32x4 gz  = { 0, 0, 0, 0 };     // groups 2/3 unused (<=2D tensor)
            i32x8 gz8 = { 0, 0, 0, 0, 0, 0, 0, 0 };
            __builtin_amdgcn_tensor_load_to_lds(g0, g1, gz, gz, gz8, 0);
        }
#elif HAVE_ASYNC
        // Each thread streams its 128B slice via async loads (ASYNCcnt-tracked).
        const __attribute__((address_space(1))) char* g =
            (const __attribute__((address_space(1))) char*)srcv + tid * 128;
        __attribute__((address_space(3))) char* l =
            (__attribute__((address_space(3))) char*)dst + tid * 128;
        #pragma unroll
        for (int j = 0; j < 8; ++j)
            __builtin_amdgcn_global_load_async_to_lds_b128(g, l, j * 16, 0);
#else
        const uint4* __restrict__ s4 = (const uint4*)srcv;
        uint4* d4 = (uint4*)dst;
        __builtin_prefetch((const char*)srcv + 32768 + tid * 64, 0, 0);
        #pragma unroll
        for (int i = 0; i < (TILE_ELEMS / 8) / 256; ++i)        // 8 x 16B per thread
            d4[tid + i * 256] = s4[tid + i * 256];
#endif
    } else {
        const v4f* __restrict__ src = (const v4f*)srcv;
        v4bf* dst4 = (v4bf*)dst;
        #pragma unroll
        for (int i = 0; i < (TILE_ELEMS / 4) / 256; ++i) {      // 16 iters
            v4f f = src[tid + i * 256];
            v4bf b;
            #pragma unroll
            for (int j = 0; j < 4; ++j) b[j] = (__bf16)f[j];
            dst4[tid + i * 256] = b;
        }
    }
}

__device__ __forceinline__ void stage_wait(int tid) {
#if HAVE_TDM
    if (tid < 32) __builtin_amdgcn_s_wait_tensorcnt(0);
#elif HAVE_ASYNC
#  if __has_builtin(__builtin_amdgcn_s_wait_asynccnt)
    __builtin_amdgcn_s_wait_asynccnt(0);
#  else
    asm volatile("s_wait_asynccnt 0x0" ::: "memory");
#  endif
#endif
}

// ---------------- pass 1: per-(head, chunk) partial attention -----------------------------
template<bool BF>
__global__ __launch_bounds__(256)
void attn_partial_kernel(const float* __restrict__ x,      // [NH][HD] current queries
                         const void* __restrict__ Kp,      // [NH][SEQ][HD]
                         const void* __restrict__ Vp,
                         float* __restrict__ partM,
                         float* __restrict__ partL,
                         float* __restrict__ partAcc) {
    __shared__ __align__(16) __bf16 qs[HD];
    __shared__ __align__(16) __bf16 tileA[TILE_ELEMS];     // 32 KB double buffer
    __shared__ __align__(16) __bf16 tileB[TILE_ELEMS];     // 32 KB
    __shared__ __align__(16) float  scores[CHUNK];
    __shared__ __align__(16) __bf16 probs[CHUNK];
    __shared__ __align__(16) float  red[256];

    const int tid  = threadIdx.x;
    const int lane = tid & 31;
    const int w    = tid >> 5;            // wave 0..7
    const int hi   = (lane >= 16);
    const int ln   = lane & 15;
    const int c    = blockIdx.x;          // chunk 0..15
    const int h    = blockIdx.y;          // head 0..31

    const float scale = 0.08838834764831844f;  // 1/sqrt(128)
    if (tid < HD) qs[tid] = (__bf16)(x[h * HD + tid] * scale);
    __syncthreads();

    const long elemSz     = BF ? 2 : 4;
    const long chunkElems = ((long)h * SEQ + (long)c * CHUNK) * HD;
    const char* Kc = (const char*)Kp + chunkElems * elemSz;
    const char* Vc = (const char*)Vp + chunkElems * elemSz;

    // ================= phase A: scores = (q*scale) . K^T =================================
    stage_issue<BF>(tileA, Kc, tid);
    for (int t = 0; t < NTILE; ++t) {
        stage_wait(tid);
        __syncthreads();                                  // tile t ready, prev compute done
        if (t + 1 < NTILE)
            stage_issue<BF>(((t + 1) & 1) ? tileB : tileA,
                            Kc + (long)(t + 1) * TILE_ELEMS * elemSz, tid);
        const __bf16* tl = (t & 1) ? tileB : tileA;

        // each wave: 16 keys' scores via 4 chained v_wmma_f32_16x16x32_bf16
        v8f cacc = {};
        const int key = w * 16 + ln;
        #pragma unroll
        for (int dc = 0; dc < 4; ++dc) {
            const int kb = dc * 32 + (hi ? 8 : 0);
            v8bf alo = *(const v8bf*)&qs[kb];
            v8bf ahi = *(const v8bf*)&qs[kb + 16];
            v16bf a = __builtin_shufflevector(alo, ahi, 0,1,2,3,4,5,6,7,8,9,10,11,12,13,14,15);
            v8bf blo = *(const v8bf*)&tl[key * HD + kb];
            v8bf bhi = *(const v8bf*)&tl[key * HD + kb + 16];
            v16bf b = __builtin_shufflevector(blo, bhi, 0,1,2,3,4,5,6,7,8,9,10,11,12,13,14,15);
            cacc = __builtin_amdgcn_wmma_f32_16x16x32_bf16(false, a, false, b, (short)0, cacc, false, false);
        }
        if (lane < 16) scores[t * TILE + w * 16 + lane] = cacc[0];
    }
    __syncthreads();

    // ================= softmax over the 512-key chunk (partial: keep m, l) ================
    float lm = -1e30f;
    #pragma unroll
    for (int i = tid; i < CHUNK; i += 256) lm = fmaxf(lm, scores[i]);
    red[tid] = lm; __syncthreads();
    #pragma unroll
    for (int s2 = 128; s2 > 0; s2 >>= 1) {
        if (tid < s2) red[tid] = fmaxf(red[tid], red[tid + s2]);
        __syncthreads();
    }
    const float m = red[0];
    __syncthreads();
    float ls = 0.f;
    #pragma unroll
    for (int i = tid; i < CHUNK; i += 256) {
        float e = __expf(scores[i] - m);
        probs[i] = (__bf16)e;
        ls += e;
    }
    red[tid] = ls; __syncthreads();
    #pragma unroll
    for (int s2 = 128; s2 > 0; s2 >>= 1) {
        if (tid < s2) red[tid] += red[tid + s2];
        __syncthreads();
    }
    if (tid == 0) { partM[h * NCHUNK + c] = m; partL[h * NCHUNK + c] = red[0]; }
    __syncthreads();

    // ================= phase C: acc = probs . V ; wave w owns dims [16w,16w+16) ===========
    v8f acc = {};
    const int dim = w * 16 + ln;
    stage_issue<BF>(tileA, Vc, tid);
    for (int t = 0; t < NTILE; ++t) {
        stage_wait(tid);
        __syncthreads();
        if (t + 1 < NTILE)
            stage_issue<BF>(((t + 1) & 1) ? tileB : tileA,
                            Vc + (long)(t + 1) * TILE_ELEMS * elemSz, tid);
        const __bf16* tl = (t & 1) ? tileB : tileA;

        #pragma unroll
        for (int s = 0; s < 4; ++s) {     // 32-key reduction subtiles
            const int pb = t * TILE + s * 32 + (hi ? 8 : 0);
            v8bf alo = *(const v8bf*)&probs[pb];
            v8bf ahi = *(const v8bf*)&probs[pb + 16];
            v16bf a = __builtin_shufflevector(alo, ahi, 0,1,2,3,4,5,6,7,8,9,10,11,12,13,14,15);
            const int keyb = s * 32 + (hi ? 8 : 0);
            v16bf b;
            #pragma unroll
            for (int j = 0; j < 8; ++j) {
                b[j]     = tl[(keyb + j)      * HD + dim];
                b[8 + j] = tl[(keyb + 16 + j) * HD + dim];
            }
            acc = __builtin_amdgcn_wmma_f32_16x16x32_bf16(false, a, false, b, (short)0, acc, false, false);
        }
    }
    if (lane < 16)
        partAcc[((long)(h * NCHUNK + c)) * HD + w * 16 + lane] = acc[0];
}

// ---------------- pass 2: combine chunk partials -> new x ---------------------------------
__global__ __launch_bounds__(HD)
void attn_combine_kernel(const float* __restrict__ partM, const float* __restrict__ partL,
                         const float* __restrict__ partAcc, float* __restrict__ xout) {
    const int h = blockIdx.x;
    const int d = threadIdx.x;
    float M = -1e30f;
    #pragma unroll
    for (int c = 0; c < NCHUNK; ++c) M = fmaxf(M, partM[h * NCHUNK + c]);
    float L = 0.f, o = 0.f;
    #pragma unroll
    for (int c = 0; c < NCHUNK; ++c) {
        float e = __expf(partM[h * NCHUNK + c] - M);
        L += partL[h * NCHUNK + c] * e;
        o += partAcc[((long)(h * NCHUNK + c)) * HD + d] * e;
    }
    xout[h * HD + d] = o / L;
}

// ---------------- launch ------------------------------------------------------------------
extern "C" void kernel_launch(void* const* d_in, const int* in_sizes, int n_in,
                              void* d_out, int out_size, void* d_ws, size_t ws_size,
                              hipStream_t stream) {
    const float* x = (const float*)d_in[0];
    const float* K = (const float*)d_in[1];
    const float* V = (const float*)d_in[2];

    const long   nKV     = (long)NH * SEQ * HD;       // 33,554,432 elements
    const size_t bfBytes = (size_t)nKV * 2;           // 64 MB per tensor

    const size_t smallBytes = ((size_t)NH * NCHUNK * 2       // partM, partL
                             + (size_t)NH * NCHUNK * HD      // partAcc
                             + (size_t)2 * NH * HD) * 4;     // x ping-pong
    const bool useBF = ws_size >= 2 * bfBytes + smallBytes;

    char* ws = (char*)d_ws;
    size_t off = 0;
    void* Kb = nullptr; void* Vb = nullptr;
    if (useBF) { Kb = ws; Vb = ws + bfBytes; off = 2 * bfBytes; }
    float* partM   = (float*)(ws + off); off += (size_t)NH * NCHUNK * 4;
    float* partL   = (float*)(ws + off); off += (size_t)NH * NCHUNK * 4;
    float* partAcc = (float*)(ws + off); off += (size_t)NH * NCHUNK * HD * 4;
    float* xb0     = (float*)(ws + off); off += (size_t)NH * HD * 4;
    float* xb1     = (float*)(ws + off);

    if (useBF)
        convert_kv_kernel<<<4096, 256, 0, stream>>>(K, V, Kb, Vb, nKV / 4);

    (void)hipMemcpyAsync(xb0, x, (size_t)NH * HD * 4, hipMemcpyDeviceToDevice, stream);

    dim3 g1(NCHUNK, NH);
    for (int l = 0; l < NLAYERS; ++l) {
        float* xin = (l & 1) ? xb1 : xb0;
        float* xo  = (l & 1) ? xb0 : xb1;
        if (useBF)
            attn_partial_kernel<true ><<<g1, 256, 0, stream>>>(xin, Kb, Vb, partM, partL, partAcc);
        else
            attn_partial_kernel<false><<<g1, 256, 0, stream>>>(xin, K,  V,  partM, partL, partAcc);
        attn_combine_kernel<<<NH, HD, 0, stream>>>(partM, partL, partAcc, xo);
    }

    // outputs: (k, v, x) concatenated flat — k/v are passthrough copies
    float* out = (float*)d_out;
    (void)hipMemcpyAsync(out,            K,   (size_t)nKV * 4,     hipMemcpyDeviceToDevice, stream);
    (void)hipMemcpyAsync(out + nKV,      V,   (size_t)nKV * 4,     hipMemcpyDeviceToDevice, stream);
    (void)hipMemcpyAsync(out + 2 * nKV,  xb0, (size_t)NH * HD * 4, hipMemcpyDeviceToDevice, stream);  // layer 31 wrote xb0
}
